// MHSA_69621419869026
// MI455X (gfx1250) — compile-verified, matrix operand order checked by reading, not run on hardware
//
#include <hip/hip_runtime.h>

typedef __bf16 v16bf __attribute__((ext_vector_type(16)));
typedef __bf16 v8bf  __attribute__((ext_vector_type(8)));
typedef float  v8f   __attribute__((ext_vector_type(8)));
typedef int    v4i_g __attribute__((vector_size(16)));   // matches builtin param type

#define S_LEN 8192
#define DMODEL 512
#define NHEAD 8
#define HDIM 64
#define RDIM 64

// ---------------------------------------------------------------------------
// CDNA5 async global->LDS copy (ASYNCcnt path), with portable fallback.
// ---------------------------------------------------------------------------
#if defined(__AMDGCN__) && __has_builtin(__builtin_amdgcn_global_load_async_to_lds_b128)
#define HAVE_ASYNC_LDS 1
#else
#define HAVE_ASYNC_LDS 0
#endif

__device__ __forceinline__ void async_copy16B(const __bf16* g, __bf16* l) {
#if HAVE_ASYNC_LDS
  __builtin_amdgcn_global_load_async_to_lds_b128(
      (__attribute__((address_space(1))) v4i_g*)(g),
      (__attribute__((address_space(3))) v4i_g*)(l), 0, 0);
#else
  *(v8bf*)l = *(const v8bf*)g;
#endif
}

__device__ __forceinline__ void async_wait_all() {
#if HAVE_ASYNC_LDS
#if __has_builtin(__builtin_amdgcn_s_wait_asynccnt)
  __builtin_amdgcn_s_wait_asynccnt(0);
#else
  asm volatile("s_wait_asynccnt 0x0" ::: "memory");
#endif
#endif
}

// ---------------------------------------------------------------------------
// WMMA helpers
// ---------------------------------------------------------------------------
__device__ __forceinline__ v8f wmma_bf16(v16bf a, v16bf b, v8f c) {
  return __builtin_amdgcn_wmma_f32_16x16x32_bf16(
      /*neg_a=*/false, a, /*neg_b=*/false, b,
      /*c_mod=*/(short)0, c, /*reuse_a=*/false, /*reuse_b=*/false);
}

// Load a 16x32 bf16 operand tile (row-major, leading dim = ld elements).
// ISA 16-bit A layout: lane L -> row = L&15, K-base = (L>=16)?8:0,
// halves 0..7 = K base+0..7, halves 8..15 = K 16+base+0..7.
__device__ __forceinline__ v16bf load_frag16x32(const __bf16* base, int ld) {
  const int lane = threadIdx.x & 31;
  const int kb = (lane >> 4) << 3;
  const __bf16* p = base + (size_t)(lane & 15) * (size_t)ld + kb;
  v8bf lo = *(const v8bf*)(p);
  v8bf hi = *(const v8bf*)(p + 16);
  v16bf f;
#pragma unroll
  for (int i = 0; i < 8; ++i) { f[i] = lo[i]; f[i + 8] = hi[i]; }
  return f;
}

// ---------------------------------------------------------------------------
// Conversion / transpose prep kernels
// ---------------------------------------------------------------------------
__global__ void k_cvt_x(const float* __restrict__ x, __bf16* __restrict__ xbf, int n) {
  int i = blockIdx.x * 256 + threadIdx.x;
  if (i < n) xbf[i] = (__bf16)x[i];
}

// WT[w][n][k] = W_w[k][n] (bf16), w in {q,k,v,p}
__global__ void k_cvt_wT(const float* __restrict__ Wq, const float* __restrict__ Wk,
                         const float* __restrict__ Wv, const float* __restrict__ Wp,
                         __bf16* __restrict__ WT) {
  int i = blockIdx.x * 256 + threadIdx.x;  // < 4*512*512
  int w = i >> 18, rem = i & 262143;
  int n = rem >> 9, k = rem & 511;
  const float* W = (w == 0) ? Wq : (w == 1) ? Wk : (w == 2) ? Wv : Wp;
  WT[(size_t)w * 262144 + (size_t)n * 512 + k] = (__bf16)W[(size_t)k * 512 + n];
}

// GT[g][r][hd] = G_g[hd][r] (bf16), g in {qG1,qG2,kG1,kG2}
__global__ void k_cvt_gT(const float* __restrict__ qG1, const float* __restrict__ qG2,
                         const float* __restrict__ kG1, const float* __restrict__ kG2,
                         __bf16* __restrict__ GT) {
  int i = blockIdx.x * 256 + threadIdx.x;  // < 4*64*64
  int g = i >> 12, rem = i & 4095;
  int r = rem >> 6, hd = rem & 63;
  const float* G = (g == 0) ? qG1 : (g == 1) ? qG2 : (g == 2) ? kG1 : kG2;
  GT[(size_t)g * 4096 + r * 64 + hd] = (__bf16)G[hd * 64 + r];
}

// ---------------------------------------------------------------------------
// QKV projection GEMM: [32768 x 512] * [512 x 512].
// Block: 32x512 output tile; 32x512 A strip async-staged into LDS (32 KB).
// Wave: 32x64 tile = acc[2][4]; each B fragment feeds 2 WMMAs.
// mode(blockIdx.y): 0=q -> Qbf[B,H,S,HD] (affine), 1=k -> Kbf, 2=v -> vT[B,H,HD,S]
// ---------------------------------------------------------------------------
__global__ __launch_bounds__(256) void k_gemm_qkv(
    const __bf16* __restrict__ xbf, const __bf16* __restrict__ WT,
    const float* __restrict__ bq, const float* __restrict__ bk, const float* __restrict__ bv,
    const float* __restrict__ gq, const float* __restrict__ betq,
    const float* __restrict__ gk, const float* __restrict__ betk,
    __bf16* __restrict__ Qbf, __bf16* __restrict__ Kbf, __bf16* __restrict__ vT) {
  __shared__ __bf16 sA[32 * DMODEL];  // 32 KB

  const int tid = threadIdx.x;
  const int mode = blockIdx.y;
  const int wave = tid >> 5;
  const int lane = tid & 31;
  const int n0 = wave * 64;
  const size_t m0 = (size_t)blockIdx.x * 32;

  {  // stage A strip: 32*512 bf16 = 32 KB, 128 B per thread
    const __bf16* g = xbf + m0 * DMODEL + (size_t)tid * 64;
    __bf16* l = sA + tid * 64;
#pragma unroll
    for (int c = 0; c < 8; ++c) async_copy16B(g + c * 8, l + c * 8);
  }
  async_wait_all();
  __syncthreads();

  const __bf16* Wb = WT + (size_t)mode * 262144 + (size_t)n0 * DMODEL;

  v8f acc[2][4] = {};
#pragma unroll 4
  for (int k0 = 0; k0 < DMODEL; k0 += 32) {
    v16bf a0 = load_frag16x32(sA + k0, DMODEL);
    v16bf a1 = load_frag16x32(sA + 16 * DMODEL + k0, DMODEL);
    v16bf b[4];
#pragma unroll
    for (int t = 0; t < 4; ++t)
      b[t] = load_frag16x32(Wb + (size_t)t * 16 * DMODEL + k0, DMODEL);
#pragma unroll
    for (int t = 0; t < 4; ++t) {
      acc[0][t] = wmma_bf16(a0, b[t], acc[0][t]);
      acc[1][t] = wmma_bf16(a1, b[t], acc[1][t]);
    }
  }

  const float* bias = (mode == 0) ? bq : (mode == 1) ? bk : bv;
  float gamma = 1.f, beta = 0.f;
  if (mode == 0) { gamma = *gq; beta = *betq; }
  else if (mode == 1) { gamma = *gk; beta = *betk; }

  const int hi8 = (lane >> 4) << 3;
  const int nl = lane & 15;
  const int bIdx = (int)(m0 >> 13);
  const int s0 = (int)(m0 & (S_LEN - 1));

#pragma unroll
  for (int mh = 0; mh < 2; ++mh) {
    const int sm = s0 + mh * 16 + hi8;
#pragma unroll
    for (int t = 0; t < 4; ++t) {
      const int n = n0 + t * 16 + nl;
      const int h = n >> 6, hd = n & 63;
      const float bs = bias[n];
      if (mode == 2) {
        v8bf pk;
#pragma unroll
        for (int j = 0; j < 8; ++j) pk[j] = (__bf16)(acc[mh][t][j] + bs);
        size_t off = (((size_t)bIdx * NHEAD + h) * HDIM + hd) * S_LEN + sm;
        *(v8bf*)(vT + off) = pk;
      } else {
        __bf16* O = (mode == 0) ? Qbf : Kbf;
#pragma unroll
        for (int j = 0; j < 8; ++j) {
          float val = gamma * (acc[mh][t][j] + bs) + beta;
          O[(((size_t)bIdx * NHEAD + h) * S_LEN + sm + j) * HDIM + hd] = (__bf16)val;
        }
      }
    }
  }
}

// ---------------------------------------------------------------------------
// Sketch: phi = (0.125 * (z@G1) * (z@G2))^2, per-wave 16x64 (full R) tile.
// G1,G2 (16 KB) are block-invariant -> async-staged into LDS.
// qk=0 -> phiQ[B,H,S,R]; qk=1 -> phiKT[B,H,R,S].
// ---------------------------------------------------------------------------
__global__ __launch_bounds__(256) void k_sketch(
    const __bf16* __restrict__ Qbf, const __bf16* __restrict__ Kbf,
    const __bf16* __restrict__ GT,
    __bf16* __restrict__ phiQ, __bf16* __restrict__ phiKT) {
  __shared__ __bf16 sG[2 * 4096];  // 16 KB: G1 then G2

  const int tid = threadIdx.x;
  const int qk = blockIdx.z;
  const int bh = blockIdx.y;
  const int wave = tid >> 5;
  const int lane = tid & 31;
  const size_t s0 = ((size_t)blockIdx.x * 8 + wave) * 16;

  {  // stage both G matrices: 8192 bf16 / 256 thr = 32 each
    const __bf16* g = GT + (size_t)(qk * 2) * 4096 + (size_t)tid * 32;
    __bf16* l = sG + tid * 32;
#pragma unroll
    for (int c = 0; c < 4; ++c) async_copy16B(g + c * 8, l + c * 8);
  }
  async_wait_all();
  __syncthreads();

  const __bf16* Z = ((qk == 0) ? Qbf : Kbf) + ((size_t)bh * S_LEN + s0) * HDIM;
  const __bf16* G1 = sG;
  const __bf16* G2 = sG + 4096;

  v8f a1[4] = {}, a2[4] = {};
#pragma unroll
  for (int k0 = 0; k0 < HDIM; k0 += 32) {
    v16bf a = load_frag16x32(Z + k0, HDIM);
#pragma unroll
    for (int t = 0; t < 4; ++t) {
      v16bf b1 = load_frag16x32(G1 + (size_t)(t * 16) * HDIM + k0, HDIM);
      v16bf b2 = load_frag16x32(G2 + (size_t)(t * 16) * HDIM + k0, HDIM);
      a1[t] = wmma_bf16(a, b1, a1[t]);
      a2[t] = wmma_bf16(a, b2, a2[t]);
    }
  }

  const int hi8 = (lane >> 4) << 3;
  const int nl = lane & 15;
#pragma unroll
  for (int t = 0; t < 4; ++t) {
    const int r = t * 16 + nl;
    if (qk == 1) {
      v8bf pk;
#pragma unroll
      for (int j = 0; j < 8; ++j) {
        float hf = 0.125f * a1[t][j] * a2[t][j];  // 1/sqrt(R)=0.125
        pk[j] = (__bf16)(hf * hf);
      }
      *(v8bf*)(phiKT + ((size_t)bh * RDIM + r) * S_LEN + s0 + hi8) = pk;
    } else {
#pragma unroll
      for (int j = 0; j < 8; ++j) {
        float hf = 0.125f * a1[t][j] * a2[t][j];
        phiQ[((size_t)bh * S_LEN + s0 + hi8 + j) * RDIM + r] = (__bf16)(hf * hf);
      }
    }
  }
}

// ---------------------------------------------------------------------------
// kv[b,h,r,d] = sum_s phi_k[s,r] * v[s,d]; one block per (b,h),
// 16 waves x one 16x16 tile, K = 8192, software-pipelined.
// ---------------------------------------------------------------------------
__global__ __launch_bounds__(512) void k_kv(const __bf16* __restrict__ phiKT,
                                            const __bf16* __restrict__ vT,
                                            float* __restrict__ kv) {
  const int bh = blockIdx.x;
  const int wave = threadIdx.x >> 5;
  const int lane = threadIdx.x & 31;
  const int r0 = (wave >> 2) * 16, d0 = (wave & 3) * 16;
  const __bf16* A = phiKT + ((size_t)bh * RDIM + r0) * S_LEN;
  const __bf16* Bp = vT + ((size_t)bh * HDIM + d0) * S_LEN;

  v8f acc = {};
  v16bf a_cur = load_frag16x32(A, S_LEN);
  v16bf b_cur = load_frag16x32(Bp, S_LEN);
#pragma unroll 4
  for (int k0 = 0; k0 < S_LEN - 32; k0 += 32) {
    v16bf a_nxt = load_frag16x32(A + k0 + 32, S_LEN);
    v16bf b_nxt = load_frag16x32(Bp + k0 + 32, S_LEN);
    acc = wmma_bf16(a_cur, b_cur, acc);
    a_cur = a_nxt;
    b_cur = b_nxt;
  }
  acc = wmma_bf16(a_cur, b_cur, acc);

  const int hi8 = (lane >> 4) << 3;
  const int nl = lane & 15;
#pragma unroll
  for (int j = 0; j < 8; ++j)
    kv[((size_t)bh * RDIM + r0 + hi8 + j) * HDIM + d0 + nl] = acc[j];
}

// ksum[b,h,r] = sum_s phi_k[s,r]  (one wave per (bh,r) row of phiKT)
__global__ __launch_bounds__(256) void k_ksum(const __bf16* __restrict__ phiKT,
                                              float* __restrict__ ksum) {
  const int row = blockIdx.x * 8 + (threadIdx.x >> 5);  // < 32*64
  const int lane = threadIdx.x & 31;
  const __bf16* p = phiKT + (size_t)row * S_LEN;
  float s = 0.f;
  for (int i = lane; i < S_LEN; i += 32) s += (float)p[i];
#pragma unroll
  for (int off = 16; off > 0; off >>= 1) s += __shfl_down(s, off, 32);
  if (lane == 0) ksum[row] = s;
}

// kvT[b,h,d,r] = bf16(kv[b,h,r,d])
__global__ void k_kvT(const float* __restrict__ kv, __bf16* __restrict__ kvT) {
  int i = blockIdx.x * 256 + threadIdx.x;  // < 32*4096
  int bh = i >> 12, rem = i & 4095, d = rem >> 6, r = rem & 63;
  kvT[((size_t)bh * HDIM + d) * RDIM + r] = (__bf16)kv[((size_t)bh * RDIM + r) * HDIM + d];
}

// num[b,h,s,d] = sum_r phi_q[s,r] * kv[r,d]; per-wave 32x64 tile, K = 64.
// kvT[bh] (8 KB) is block-invariant -> async-staged into LDS.
__global__ __launch_bounds__(256) void k_num(const __bf16* __restrict__ phiQ,
                                             const __bf16* __restrict__ kvT,
                                             float* __restrict__ numb) {
  __shared__ __bf16 sB[4096];  // 8 KB

  const int tid = threadIdx.x;
  const int bh = blockIdx.x >> 5;                      // 32 blocks per (b,h)
  const int mt = (blockIdx.x * 8 + (tid >> 5)) & 255;  // 32-row tile within (b,h)
  const int lane = tid & 31;
  const size_t s0 = (size_t)mt * 32;

  {  // stage kvT[bh]: 4096 bf16 / 256 thr = 16 each
    const __bf16* g = kvT + (size_t)bh * 4096 + (size_t)tid * 16;
    __bf16* l = sB + tid * 16;
#pragma unroll
    for (int c = 0; c < 2; ++c) async_copy16B(g + c * 8, l + c * 8);
  }
  async_wait_all();
  __syncthreads();

  const __bf16* A = phiQ + ((size_t)bh * S_LEN + s0) * RDIM;

  v8f acc[2][4] = {};
#pragma unroll
  for (int k0 = 0; k0 < RDIM; k0 += 32) {
    v16bf a0 = load_frag16x32(A + k0, RDIM);
    v16bf a1 = load_frag16x32(A + (size_t)16 * RDIM + k0, RDIM);
#pragma unroll
    for (int t = 0; t < 4; ++t) {
      v16bf b = load_frag16x32(sB + (size_t)(t * 16) * RDIM + k0, RDIM);
      acc[0][t] = wmma_bf16(a0, b, acc[0][t]);
      acc[1][t] = wmma_bf16(a1, b, acc[1][t]);
    }
  }
  const int hi8 = (lane >> 4) << 3;
  const int nl = lane & 15;
#pragma unroll
  for (int mh = 0; mh < 2; ++mh)
#pragma unroll
    for (int t = 0; t < 4; ++t)
#pragma unroll
      for (int j = 0; j < 8; ++j)
        numb[((size_t)bh * S_LEN + s0 + mh * 16 + hi8 + j) * HDIM + t * 16 + nl] =
            acc[mh][t][j];
}

// den = phi_q . ksum + eps ; attn[b,s,h*64+d] = bf16(num/den)
__global__ void k_norm(const __bf16* __restrict__ phiQ, const float* __restrict__ ksum,
                       const float* __restrict__ numb, __bf16* __restrict__ attnbf) {
  int i = blockIdx.x * 256 + threadIdx.x;  // < 32*8192 (bh*S + s)
  int bh = i >> 13, s = i & (S_LEN - 1);
  int b = bh >> 3, h = bh & 7;
  const __bf16* pq = phiQ + (size_t)i * RDIM;
  const float* ks = ksum + bh * RDIM;
  float den = 1e-6f;
#pragma unroll 8
  for (int r = 0; r < RDIM; ++r) den += (float)pq[r] * ks[r];
  float inv = 1.f / den;
  const float* nm = numb + (size_t)i * HDIM;
  __bf16* ob = attnbf + ((size_t)b * S_LEN + s) * DMODEL + h * HDIM;
#pragma unroll 8
  for (int d = 0; d < HDIM; ++d) ob[d] = (__bf16)(nm[d] * inv);
}

// Output projection: attn[32768x512] * Wp + bp -> fp32 out.
// Same 32-row LDS A-staging as k_gemm_qkv.
__global__ __launch_bounds__(256) void k_gemm_out(const __bf16* __restrict__ attnbf,
                                                  const __bf16* __restrict__ WpT,
                                                  const float* __restrict__ bp,
                                                  float* __restrict__ out) {
  __shared__ __bf16 sA[32 * DMODEL];  // 32 KB

  const int tid = threadIdx.x;
  const int wave = tid >> 5;
  const int lane = tid & 31;
  const int n0 = wave * 64;
  const size_t m0 = (size_t)blockIdx.x * 32;

  {
    const __bf16* g = attnbf + m0 * DMODEL + (size_t)tid * 64;
    __bf16* l = sA + tid * 64;
#pragma unroll
    for (int c = 0; c < 8; ++c) async_copy16B(g + c * 8, l + c * 8);
  }
  async_wait_all();
  __syncthreads();

  const __bf16* Wb = WpT + (size_t)n0 * DMODEL;

  v8f acc[2][4] = {};
#pragma unroll 4
  for (int k0 = 0; k0 < DMODEL; k0 += 32) {
    v16bf a0 = load_frag16x32(sA + k0, DMODEL);
    v16bf a1 = load_frag16x32(sA + 16 * DMODEL + k0, DMODEL);
    v16bf b[4];
#pragma unroll
    for (int t = 0; t < 4; ++t)
      b[t] = load_frag16x32(Wb + (size_t)t * 16 * DMODEL + k0, DMODEL);
#pragma unroll
    for (int t = 0; t < 4; ++t) {
      acc[0][t] = wmma_bf16(a0, b[t], acc[0][t]);
      acc[1][t] = wmma_bf16(a1, b[t], acc[1][t]);
    }
  }

  const int hi8 = (lane >> 4) << 3;
  const int nl = lane & 15;
#pragma unroll
  for (int mh = 0; mh < 2; ++mh)
#pragma unroll
    for (int t = 0; t < 4; ++t) {
      const int n = n0 + t * 16 + nl;
      const float bs = bp[n];
#pragma unroll
      for (int j = 0; j < 8; ++j)
        out[(m0 + mh * 16 + hi8 + j) * DMODEL + n] = acc[mh][t][j] + bs;
    }
}

// ---------------------------------------------------------------------------
extern "C" void kernel_launch(void* const* d_in, const int* in_sizes, int n_in,
                              void* d_out, int out_size, void* d_ws, size_t ws_size,
                              hipStream_t stream) {
  (void)in_sizes; (void)n_in; (void)out_size; (void)ws_size;
  const float* x   = (const float*)d_in[0];
  const float* Wq  = (const float*)d_in[1];
  const float* bq  = (const float*)d_in[2];
  const float* Wk  = (const float*)d_in[3];
  const float* bk  = (const float*)d_in[4];
  const float* Wv  = (const float*)d_in[5];
  const float* bv  = (const float*)d_in[6];
  const float* Wp  = (const float*)d_in[7];
  const float* bp  = (const float*)d_in[8];
  const float* gq  = (const float*)d_in[9];
  const float* btq = (const float*)d_in[10];
  const float* gk  = (const float*)d_in[11];
  const float* btk = (const float*)d_in[12];
  const float* qG1 = (const float*)d_in[13];
  const float* qG2 = (const float*)d_in[14];
  const float* kG1 = (const float*)d_in[15];
  const float* kG2 = (const float*)d_in[16];

  char* w = (char*)d_ws;
  __bf16* xbf   = (__bf16*)w;  w += (size_t)33554432;   // 32768*512 bf16
  __bf16* WT    = (__bf16*)w;  w += (size_t)2097152;    // 4*512*512 bf16
  __bf16* GT    = (__bf16*)w;  w += (size_t)32768;      // 4*64*64 bf16
  __bf16* Qbf   = (__bf16*)w;  w += (size_t)33554432;   // [B,H,S,HD] bf16
  __bf16* Kbf   = (__bf16*)w;  w += (size_t)33554432;
  __bf16* vT    = (__bf16*)w;  w += (size_t)33554432;   // [B,H,HD,S] bf16
  __bf16* phiQ  = (__bf16*)w;  w += (size_t)33554432;   // [B,H,S,R] bf16
  __bf16* phiKT = (__bf16*)w;  w += (size_t)33554432;   // [B,H,R,S] bf16
  float*  kv    = (float*)w;   w += (size_t)524288;     // [B,H,R,HD] f32
  __bf16* kvT   = (__bf16*)w;  w += (size_t)262144;     // [B,H,HD,R] bf16
  float*  ksum  = (float*)w;   w += (size_t)8192;       // [B,H,R] f32
  float*  numb  = (float*)w;   w += (size_t)67108864;   // [B,H,S,HD] f32
  __bf16* attn  = (__bf16*)w;  w += (size_t)33554432;   // [B,S,D] bf16

  k_cvt_x  <<<65536, 256, 0, stream>>>(x, xbf, 32768 * 512);
  k_cvt_wT <<<4096, 256, 0, stream>>>(Wq, Wk, Wv, Wp, WT);
  k_cvt_gT <<<64, 256, 0, stream>>>(qG1, qG2, kG1, kG2, GT);

  k_gemm_qkv<<<dim3(1024, 3), 256, 0, stream>>>(xbf, WT, bq, bk, bv,
                                                gq, btq, gk, btk, Qbf, Kbf, vT);
  k_sketch  <<<dim3(64, 32, 2), 256, 0, stream>>>(Qbf, Kbf, GT, phiQ, phiKT);
  k_kv      <<<32, 512, 0, stream>>>(phiKT, vT, kv);
  k_ksum    <<<256, 256, 0, stream>>>(phiKT, ksum);
  k_kvT     <<<512, 256, 0, stream>>>(kv, kvT);
  k_num     <<<1024, 256, 0, stream>>>(phiQ, kvT, numb);
  k_norm    <<<1024, 256, 0, stream>>>(phiQ, ksum, numb, attn);
  k_gemm_out<<<1024, 256, 0, stream>>>(attn, WT + 3 * 262144, bp, (float*)d_out);
}
